// MyCNN_30786325578327
// MI455X (gfx1250) — compile-verified, matrix operand order checked by reading, not run on hardware
//
#include <hip/hip_runtime.h>

typedef __attribute__((ext_vector_type(16))) _Float16 v16h;
typedef __attribute__((ext_vector_type(8)))  float    v8f;

#define EPSF   1e-5f
#define BATCH  32768
#define NPOS   196      // 14*14 conv outputs per image per channel
#define KDIM   392      // 8*7*7 flattened features
#define KPAD   416      // 13 * 32 (WMMA K tiles)
#define KTILES 13
#define FST    416      // LDS feature row stride (halves)

// ---------------------------------------------------------------------------
// Kernel 0: zero the stats accumulators (ws[0..15]) so atomics start clean.
// ---------------------------------------------------------------------------
__global__ __launch_bounds__(64) void k_init(float* __restrict__ ws) {
    int t = threadIdx.x;
    if (t < 16) ws[t] = 0.0f;
}

// ---------------------------------------------------------------------------
// Kernel 1: conv (recomputed, nothing stored) -> per-channel sum/sumsq.
// One 3x3 input patch feeds all 8 output channels (single input channel).
// Block-level LDS reduction, then 16 global float atomics per block.
// ---------------------------------------------------------------------------
__global__ __launch_bounds__(256) void k_conv_stats(
    const float* __restrict__ x, const float* __restrict__ Wc,
    const float* __restrict__ bc, float* __restrict__ ws) {
    __shared__ float red[16];
    int t = threadIdx.x;
    if (t < 16) red[t] = 0.0f;
    __syncthreads();

    float sum[8], ssq[8];
#pragma unroll
    for (int c = 0; c < 8; ++c) { sum[c] = 0.0f; ssq[c] = 0.0f; }

    const long total  = (long)BATCH * NPOS;
    const long stride = (long)gridDim.x * blockDim.x;
    for (long idx = (long)blockIdx.x * blockDim.x + t; idx < total; idx += stride) {
        int i   = (int)(idx / NPOS);
        int pos = (int)(idx - (long)i * NPOS);
        int oy  = pos / 14, ox = pos - oy * 14;
        const float* xi = x + (long)i * 784;

        float p[9];
#pragma unroll
        for (int ky = 0; ky < 3; ++ky) {
            int iy = 2 * oy - 1 + ky;
#pragma unroll
            for (int kx = 0; kx < 3; ++kx) {
                int ix = 2 * ox - 1 + kx;
                bool ok = (iy >= 0) & (iy < 28) & (ix >= 0) & (ix < 28);
                p[ky * 3 + kx] = ok ? xi[iy * 28 + ix] : 0.0f;
            }
        }
#pragma unroll
        for (int c = 0; c < 8; ++c) {
            float v = bc[c];
#pragma unroll
            for (int k = 0; k < 9; ++k) v = fmaf(p[k], Wc[c * 9 + k], v);
            sum[c] += v;
            ssq[c] += v * v;
        }
    }
#pragma unroll
    for (int c = 0; c < 8; ++c) {
        atomicAdd(&red[c],     sum[c]);
        atomicAdd(&red[8 + c], ssq[c]);
    }
    __syncthreads();
    if (t < 16) atomicAdd(&ws[t], red[t]);
}

// ---------------------------------------------------------------------------
// Kernel 2: finalize BN scale/shift (ws[16..31]) and build the f16 B-panel
// PRE-SWIZZLED into the WMMA B-fragment register layout:
//   wsB[(kt*32 + lane)*16 + j] = Wfc[n][k]   (zero-padded)
//   with n = lane&15,  k = kt*32 + 16*(j>=8) + 8*(lane>=16) + (j&7)
// so phase B loads one aligned 32-byte v16h per lane per k-tile
// (a wave reads one contiguous 1KB segment -> 2x global_load_b128/lane).
// ---------------------------------------------------------------------------
__global__ __launch_bounds__(256) void k_finalize(
    const float* __restrict__ gamma, const float* __restrict__ beta,
    const float* __restrict__ Wfc, float* __restrict__ ws,
    _Float16* __restrict__ wsB) {
    int t = threadIdx.x;
    if (t < 8) {
        float n    = (float)BATCH * (float)NPOS;
        float mean = ws[t] / n;
        float var  = ws[8 + t] / n - mean * mean;
        float sc   = gamma[t] * rsqrtf(var + EPSF);
        ws[16 + t] = sc;
        ws[24 + t] = beta[t] - sc * mean;   // BN folded: y = sc*conv + shift
    }
    for (int idx = t; idx < KTILES * 32 * 16; idx += blockDim.x) {
        int j    = idx & 15;
        int lane = (idx >> 4) & 31;
        int kt   = idx >> 9;
        int nn   = lane & 15;
        int k    = kt * 32 + ((j >> 3) * 16) + ((lane >> 4) * 8) + (j & 7);
        float v  = (k < KDIM && nn < 10) ? Wfc[nn * KDIM + k] : 0.0f;
        wsB[idx] = (_Float16)v;
    }
}

// ---------------------------------------------------------------------------
// Kernel 3: fused conv(+folded BN)+ReLU+maxpool into LDS f16 features,
// then FC via v_wmma_f32_16x16x32_f16. 128 threads (4 waves) / 64 images;
// each wave owns a 16-image M-tile, K = 416 = 13 WMMA steps (unrolled).
// x is read again here but is L2-resident (103 MB < 192 MB L2).
// ---------------------------------------------------------------------------
__global__ __launch_bounds__(128) void k_fused(
    const float* __restrict__ x,  const float* __restrict__ Wc,
    const float* __restrict__ bc, const float* __restrict__ bfc,
    const float* __restrict__ ws, const _Float16* __restrict__ wsB,
    float* __restrict__ out) {
    __shared__ _Float16 feat[64 * FST];      // 53248 B
    int t = threadIdx.x;
    int blkImg = blockIdx.x * 64;

    // zero the K padding columns (392..415) of every feature row
    for (int idx = t; idx < 64 * (KPAD - KDIM); idx += 128) {
        int r = idx / (KPAD - KDIM);
        int c = idx - r * (KPAD - KDIM);
        feat[r * FST + KDIM + c] = (_Float16)0.0f;
    }

    // Phase A: pooled features. Each item = one (image, pooled-pos) pair,
    // producing all 8 channels from one shared 5x5 input patch.
    for (int idx = t; idx < 64 * 49; idx += 128) {
        int il = idx / 49;
        int p  = idx - il * 49;
        int py = p / 7, px = p - py * 7;
        const float* xi = x + (long)(blkImg + il) * 784;

        float patch[25];
        int iy0 = 4 * py - 1, ix0 = 4 * px - 1;
#pragma unroll
        for (int r = 0; r < 5; ++r) {
            int iy = iy0 + r;
#pragma unroll
            for (int c = 0; c < 5; ++c) {
                int ix = ix0 + c;
                bool ok = (iy >= 0) & (iy < 28) & (ix >= 0) & (ix < 28);
                patch[r * 5 + c] = ok ? xi[iy * 28 + ix] : 0.0f;
            }
        }
#pragma unroll
        for (int c = 0; c < 8; ++c) {
            float sc = ws[16 + c], sh = ws[24 + c], bb = bc[c];
            float mx = -3.0e38f;
#pragma unroll
            for (int dy = 0; dy < 2; ++dy) {
#pragma unroll
                for (int dx = 0; dx < 2; ++dx) {
                    float v = bb;
#pragma unroll
                    for (int ky = 0; ky < 3; ++ky)
#pragma unroll
                        for (int kx = 0; kx < 3; ++kx)
                            v = fmaf(patch[(2 * dy + ky) * 5 + (2 * dx + kx)],
                                     Wc[c * 9 + ky * 3 + kx], v);
                    v = fmaf(sc, v, sh);        // folded batchnorm
                    mx = fmaxf(mx, v);          // maxpool (relu after: monotone)
                }
            }
            mx = fmaxf(mx, 0.0f);               // relu
            feat[il * FST + c * 49 + p] = (_Float16)mx;   // torch NCHW flatten
        }
    }
    __syncthreads();

    // Phase B: FC as WMMA. ISA 7.12.2 wave32 layouts:
    //   A: lane&15 = M row; elem j <-> K = 16*(j>=8) + 8*(lane>=16) + (j&7)
    //   B: pre-swizzled in wsB -> direct aligned v16h load per lane
    //   C/D: VGPR r <-> M = r + 8*(lane>=16); lane&15 = N
    int wave = t >> 5, lane = t & 31;
    int nn = lane & 15;
    int hi = (lane >> 4) * 8;
    const _Float16* arow = feat + (wave * 16 + nn) * FST;
    const v16h* bpanel = (const v16h*)wsB;   // 32B-aligned fragments

    v8f acc = {};
#pragma unroll
    for (int kt = 0; kt < KTILES; ++kt) {
        const _Float16* ap = arow + kt * 32;
        v16h a;
#pragma unroll
        for (int j = 0; j < 8; ++j) {
            a[j]     = ap[hi + j];
            a[8 + j] = ap[16 + hi + j];
        }
        v16h b = bpanel[kt * 32 + lane];     // 2x global_load_b128, coalesced
        acc = __builtin_amdgcn_wmma_f32_16x16x32_f16(
            /*neg_a=*/false, a, /*neg_b=*/false, b,
            /*c_mod=*/(short)0, acc, /*reuse_a=*/false, /*reuse_b=*/false);
    }

    if (nn < 10) {
        float bias = bfc[nn];
        int rbase = blkImg + wave * 16 + ((lane >> 4) ? 8 : 0);
#pragma unroll
        for (int r = 0; r < 8; ++r)
            out[(long)(rbase + r) * 10 + nn] = acc[r] + bias;
    }
}

// ---------------------------------------------------------------------------
extern "C" void kernel_launch(void* const* d_in, const int* in_sizes, int n_in,
                              void* d_out, int out_size, void* d_ws, size_t ws_size,
                              hipStream_t stream) {
    const float* x     = (const float*)d_in[0];
    const float* Wc    = (const float*)d_in[1];
    const float* bc    = (const float*)d_in[2];
    const float* gamma = (const float*)d_in[3];
    const float* beta  = (const float*)d_in[4];
    const float* Wfc   = (const float*)d_in[5];
    const float* bfc   = (const float*)d_in[6];

    float*    ws  = (float*)d_ws;                                  // [0..15] stats, [16..31] scale/shift
    _Float16* wsB = (_Float16*)((char*)d_ws + 64 * sizeof(float)); // swizzled f16 B-panel
    float*    out = (float*)d_out;

    k_init      <<<1, 64, 0, stream>>>(ws);
    k_conv_stats<<<1024, 256, 0, stream>>>(x, Wc, bc, ws);
    k_finalize  <<<1, 256, 0, stream>>>(gamma, beta, Wfc, ws, wsB);
    k_fused     <<<BATCH / 64, 128, 0, stream>>>(x, Wc, bc, bfc, ws, wsB, out);
}
